// MGGAT_58634893525376
// MI455X (gfx1250) — matrix operand order repaired and checked by reading, not compiled, not run on hardware
//
#include <hip/hip_runtime.h>
#include <cstdint>
#include <cstddef>

// ---------------------------------------------------------------------------
// Types for CDNA5 WMMA (wave32): 16x16x32 f16 -> f32
// ---------------------------------------------------------------------------
typedef __attribute__((ext_vector_type(16))) _Float16 v16h;
typedef __attribute__((ext_vector_type(8)))  _Float16 v8h;
typedef __attribute__((ext_vector_type(4)))  _Float16 v4h;
typedef __attribute__((ext_vector_type(8)))  float    v8f;

__device__ __forceinline__ v8f wmma16(v16h a, v16h b, v8f c) {
  return __builtin_amdgcn_wmma_f32_16x16x32_f16(
      /*neg_a=*/false, a, /*neg_b=*/false, b,
      /*c_mod=*/(short)0, c, /*reuse_a=*/false, /*reuse_b=*/false);
}

__device__ __forceinline__ v16h mkfrag(v8h lo, v8h hi) {
  return __builtin_shufflevector(lo, hi, 0, 1, 2, 3, 4, 5, 6, 7,
                                 8, 9, 10, 11, 12, 13, 14, 15);
}

// Ordered-uint mapping for float atomic max
__device__ __forceinline__ unsigned ford(float f) {
  unsigned u = __float_as_uint(f);
  return (u & 0x80000000u) ? ~u : (u | 0x80000000u);
}
__device__ __forceinline__ float fuord(unsigned u) {
  return (u & 0x80000000u) ? __uint_as_float(u & 0x7fffffffu)
                           : __uint_as_float(~u);
}

#define EMPTYKEY 0xFFFFFFFFFFFFFFFFull
#define HBITS 22u
#define HSIZE (1u << HBITS)

__device__ __forceinline__ unsigned hashKey(unsigned long long k) {
  k ^= k >> 33; k *= 0xff51afd7ed558ccdULL;
  k ^= k >> 33; k *= 0xc4ceb9fe1a85ec53ULL;
  k ^= k >> 33;
  return (unsigned)k & (HSIZE - 1u);
}

// ---------------------------------------------------------------------------
// Fill kernels (d_ws is poisoned -> must init everything we accumulate into)
// ---------------------------------------------------------------------------
__global__ void k_fill_f32(float* p, float v, long long n) {
  long long i = (long long)blockIdx.x * blockDim.x + threadIdx.x;
  if (i < n) p[i] = v;
}
__global__ void k_fill_u32(unsigned* p, unsigned v, long long n) {
  long long i = (long long)blockIdx.x * blockDim.x + threadIdx.x;
  if (i < n) p[i] = v;
}
__global__ void k_fill_u64(unsigned long long* p, unsigned long long v, long long n) {
  long long i = (long long)blockIdx.x * blockDim.x + threadIdx.x;
  if (i < n) p[i] = v;
}

// ---------------------------------------------------------------------------
// Pack weights [K x (Na|Nb)] (two column blocks) into WMMA B-fragment layout:
// element ((kt*CT+ct)*32 + lane)*16 + j  <-  W[kt*32 + (lane>>4)*8 + (j&7) + (j>>3)*16][ct*16 + (lane&15)]
// Each wave then fetches its B fragment as one contiguous 32B load (L2-hot).
// ---------------------------------------------------------------------------
__global__ void k_pack_b(const float* __restrict__ Wa, int Na,
                         const float* __restrict__ Wb, int Nb,
                         _Float16* __restrict__ out, int K) {
  const int N = Na + Nb;
  const int CT = N / 16;
  int idx = blockIdx.x * blockDim.x + threadIdx.x;
  int total = (K / 32) * CT * 32 * 16;
  if (idx >= total) return;
  int j = idx & 15;
  int lane = (idx >> 4) & 31;
  int tile = idx >> 9;                 // kt*CT + ct
  int kt = tile / CT, ct = tile - kt * CT;
  int k = kt * 32 + (lane >> 4) * 8 + (j & 7) + (j >> 3) * 16;
  int c = ct * 16 + (lane & 15);
  float v = (c < Na) ? Wa[(size_t)k * Na + c] : Wb[(size_t)k * Nb + (c - Na)];
  out[idx] = (_Float16)v;
}

// ---------------------------------------------------------------------------
// Fused input GEMM: [M,256] x [256,192] -> H1 [M,64], P [M,128]
// Per wave: rt = wave&3 is constant -> one A fragment per K-step (2x ds_load_b128);
// 6 B fragments off a single base pointer with immediate offsets (stride 2048B),
// then 6 back-to-back WMMAs on independent accumulators.
// ---------------------------------------------------------------------------
__global__ __launch_bounds__(256) void k_gemm1(
    const float* __restrict__ S, const _Float16* __restrict__ Bpack,
    float* __restrict__ H1, float* __restrict__ P, int M)
{
  __shared__ _Float16 As[64][40];      // 64 x 32, row stride 80B (16B multiple)
  const int tid = threadIdx.x;
  const int wave = tid >> 5, lane = tid & 31;
  const int rowBlk = blockIdx.x * 64;
  const int lr = lane & 15, lh = lane >> 4;
  const int rt = wave & 3;             // constant row tile per wave
  const int ct0 = wave >> 2;           // col tiles: ct0 + 2*t, t=0..5
  const v16h* Bp = (const v16h*)Bpack;
  v8f acc[6] = {};

  for (int kt = 0; kt < 8; ++kt) {
    const int k0 = kt * 32;
    // stage A tile: f32 -> f16, vectorized (float4 load, 4-half store)
    for (int i = tid; i < 64 * 8; i += 256) {
      int r = i >> 3, c = (i & 7) * 4;
      int gr = rowBlk + r;
      float4 v = make_float4(0.f, 0.f, 0.f, 0.f);
      if (gr < M) {
        v = *(const float4*)&S[(size_t)gr * 256 + k0 + c];
        if (kt < 7) __builtin_prefetch(&S[(size_t)gr * 256 + k0 + 32 + c], 0, 3);
      }
      v4h h = {(_Float16)v.x, (_Float16)v.y, (_Float16)v.z, (_Float16)v.w};
      *(v4h*)&As[r][c] = h;
    }
    __syncthreads();
    const _Float16* ap = &As[rt * 16 + lr][lh * 8];
    v16h a = mkfrag(*(const v8h*)ap, *(const v8h*)(ap + 16));
    const v16h* bb = Bp + (size_t)(kt * 12 + ct0) * 32 + lane;
    v16h b0 = bb[0],   b1 = bb[64],  b2 = bb[128];
    v16h b3 = bb[192], b4 = bb[256], b5 = bb[320];
    acc[0] = wmma16(a, b0, acc[0]);
    acc[1] = wmma16(a, b1, acc[1]);
    acc[2] = wmma16(a, b2, acc[2]);
    acc[3] = wmma16(a, b3, acc[3]);
    acc[4] = wmma16(a, b4, acc[4]);
    acc[5] = wmma16(a, b5, acc[5]);
    __syncthreads();
  }
#pragma unroll
  for (int t = 0; t < 6; ++t) {
    int col = (ct0 + 2 * t) * 16 + lr;
#pragma unroll
    for (int v = 0; v < 8; ++v) {
      int row = rowBlk + rt * 16 + lh * 8 + v;
      if (row < M) {
        if (col < 64) H1[(size_t)row * 64 + col] = acc[t][v];
        else          P[(size_t)row * 128 + (col - 64)] = acc[t][v];
      }
    }
  }
}

// el = H1 @ a[:64], er = H1 @ a[64:]
__global__ void k_elr(const float* __restrict__ H1, const float* __restrict__ av,
                      float* __restrict__ el, float* __restrict__ er, int M) {
  int i = blockIdx.x * blockDim.x + threadIdx.x;
  if (i >= M) return;
  const float* h = H1 + (size_t)i * 64;
  float sl = 0.f, sr = 0.f;
#pragma unroll 8
  for (int k = 0; k < 64; ++k) { float x = h[k]; sl += x * av[k]; sr += x * av[64 + k]; }
  el[i] = sl; er[i] = sr;
}

// ---------------------------------------------------------------------------
// User graph segment softmax (3 passes) + scatter aggregation
// ---------------------------------------------------------------------------
__global__ void k_edge_pass1(const int* __restrict__ ei, int E,
                             const float* __restrict__ el, const float* __restrict__ er,
                             float* __restrict__ es, unsigned* __restrict__ mx) {
  int i = blockIdx.x * blockDim.x + threadIdx.x;
  if (i >= E) return;
  int src = ei[i], dst = ei[E + i];
  float e = el[dst] + er[src];
  e = (e >= 0.f) ? e : 0.2f * e;          // leaky relu
  es[i] = e;
  atomicMax(&mx[dst], ford(e));
}
__global__ void k_edge_pass2(const int* __restrict__ ei, int E,
                             float* __restrict__ es, const unsigned* __restrict__ mx,
                             float* __restrict__ sm) {
  int i = blockIdx.x * blockDim.x + threadIdx.x;
  if (i >= E) return;
  int dst = ei[E + i];
  float ex = expf(es[i] - fuord(mx[dst]));
  es[i] = ex;
  atomicAdd(&sm[dst], ex);
}
__global__ void k_edge_scatter(const int* __restrict__ ei, int E,
                               const float* __restrict__ es, const float* __restrict__ sm,
                               const float* __restrict__ H1, float* __restrict__ H2) {
  long long t = (long long)blockIdx.x * blockDim.x + threadIdx.x;
  if (t >= (long long)E * 64) return;
  int e = (int)(t >> 6), f = (int)(t & 63);
  int src = ei[e], dst = ei[E + e];
  float alpha = es[e] / (sm[dst] + 1e-16f);
  atomicAdd(&H2[(size_t)dst * 64 + f], alpha * H1[(size_t)src * 64 + f]);
}

// ---------------------------------------------------------------------------
// Biz multi-graph: hash-based duplicate-edge merge (== lexsort + group sum)
// ---------------------------------------------------------------------------
__global__ void k_hash_insert(const int* __restrict__ e0, const int* __restrict__ e1,
                              const int* __restrict__ e2, int E,
                              const float* __restrict__ el, const float* __restrict__ er,
                              const float* __restrict__ omega,
                              unsigned long long* __restrict__ keys,
                              float* __restrict__ vals) {
  int i = blockIdx.x * blockDim.x + threadIdx.x;
  if (i >= 3 * E) return;
  int g = i / E, j = i - g * E;
  const int* ei = (g == 0) ? e0 : ((g == 1) ? e1 : e2);
  int src = ei[j], dst = ei[E + j];
  float o0 = omega[0], o1 = omega[1], o2 = omega[2];
  float m = fmaxf(o0, fmaxf(o1, o2));
  float x0 = expf(o0 - m), x1 = expf(o1 - m), x2 = expf(o2 - m);
  float om = ((g == 0) ? x0 : (g == 1) ? x1 : x2) / (x0 + x1 + x2);
  float e = om * (el[dst] + er[src]);
  unsigned long long key =
      ((unsigned long long)(unsigned)src << 32) | (unsigned)dst;
  unsigned h = hashKey(key);
  for (;;) {
    unsigned long long prev = atomicCAS(&keys[h], EMPTYKEY, key);
    if (prev == EMPTYKEY || prev == key) { atomicAdd(&vals[h], e); break; }
    h = (h + 1u) & (HSIZE - 1u);
  }
}
__global__ void k_hash_max(const unsigned long long* __restrict__ keys,
                           float* __restrict__ vals, unsigned* __restrict__ mx) {
  unsigned i = blockIdx.x * blockDim.x + threadIdx.x;
  if (i >= HSIZE) return;
  unsigned long long key = keys[i];
  if (key == EMPTYKEY) return;
  float e = vals[i];
  e = (e >= 0.f) ? e : 0.2f * e;          // leaky(merged score)
  vals[i] = e;
  atomicMax(&mx[(unsigned)(key & 0xffffffffu)], ford(e));
}
__global__ void k_hash_exp(const unsigned long long* __restrict__ keys,
                           float* __restrict__ vals, const unsigned* __restrict__ mx,
                           float* __restrict__ sm) {
  unsigned i = blockIdx.x * blockDim.x + threadIdx.x;
  if (i >= HSIZE) return;
  unsigned long long key = keys[i];
  if (key == EMPTYKEY) return;
  unsigned dst = (unsigned)(key & 0xffffffffu);
  float ex = expf(vals[i] - fuord(mx[dst]));
  vals[i] = ex;
  atomicAdd(&sm[dst], ex);
}
__global__ void k_hash_scatter(const unsigned long long* __restrict__ keys,
                               const float* __restrict__ vals,
                               const float* __restrict__ sm,
                               const float* __restrict__ H1, float* __restrict__ H2) {
  unsigned i = blockIdx.x * blockDim.x + threadIdx.x;
  if (i >= HSIZE) return;
  unsigned long long key = keys[i];
  if (key == EMPTYKEY) return;
  int src = (int)(key >> 32), dst = (int)(key & 0xffffffffu);
  float alpha = vals[i] / (sm[dst] + 1e-16f);
  const float* hs = H1 + (size_t)src * 64;
  float* hd = H2 + (size_t)dst * 64;
#pragma unroll 8
  for (int f = 0; f < 64; ++f) atomicAdd(&hd[f], alpha * hs[f]);
}

// ---------------------------------------------------------------------------
// Fused layer 2: H3 = elu(H2@W2 + P + b1) (kept in LDS as f16),
// OUT = relu(H3@W3) + H4. Both B operands pre-packed in global (L2-hot).
// rt = wave&3 constant -> hoisted A fragments; B via base + immediate offsets.
// ---------------------------------------------------------------------------
__global__ __launch_bounds__(256) void k_layer2(
    const float* __restrict__ H2, const float* __restrict__ P,
    const float* __restrict__ b1, const _Float16* __restrict__ W2pack,
    const _Float16* __restrict__ W3pack, const float* __restrict__ H4,
    float* __restrict__ OUT, int M)
{
  __shared__ _Float16 As[64][72];      // H2 tile [64x64], row stride 144B
  __shared__ _Float16 H3s[64][136];    // [64x128], row stride 272B
  const int tid = threadIdx.x, wave = tid >> 5, lane = tid & 31;
  const int rowBlk = blockIdx.x * 64;
  const int lr = lane & 15, lh = lane >> 4;
  const int rt = wave & 3;             // constant row tile per wave
  const int ct0 = wave >> 2;           // 0 or 1
  const v16h* W2p = (const v16h*)W2pack;   // tiles: kt(2) x ct(8)
  const v16h* W3p = (const v16h*)W3pack;   // tiles: kt(4) x ct(4)

  for (int i = tid; i < 64 * 16; i += 256) {
    int r = i >> 4, c = (i & 15) * 4;
    int gr = rowBlk + r;
    float4 v = make_float4(0.f, 0.f, 0.f, 0.f);
    if (gr < M) v = *(const float4*)&H2[(size_t)gr * 64 + c];
    v4h h = {(_Float16)v.x, (_Float16)v.y, (_Float16)v.z, (_Float16)v.w};
    *(v4h*)&As[r][c] = h;
  }
  __syncthreads();

  // ---- stage 1: 4 col-tiles per wave (ct = ct0 + 2t), K = 64 (kt = 0,1) ----
  {
    const _Float16* ap = &As[rt * 16 + lr][lh * 8];
    v16h a0 = mkfrag(*(const v8h*)ap, *(const v8h*)(ap + 16));
    v16h a1 = mkfrag(*(const v8h*)(ap + 32), *(const v8h*)(ap + 48));
    const v16h* bb = W2p + (size_t)ct0 * 32 + lane;   // ct stride 2*32 = 64, kt stride 8*32 = 256
    v8f acc[4] = {};
    acc[0] = wmma16(a0, bb[0],   acc[0]);
    acc[1] = wmma16(a0, bb[64],  acc[1]);
    acc[2] = wmma16(a0, bb[128], acc[2]);
    acc[3] = wmma16(a0, bb[192], acc[3]);
    acc[0] = wmma16(a1, bb[256], acc[0]);
    acc[1] = wmma16(a1, bb[320], acc[1]);
    acc[2] = wmma16(a1, bb[384], acc[2]);
    acc[3] = wmma16(a1, bb[448], acc[3]);
#pragma unroll
    for (int t = 0; t < 4; ++t) {
      int col = (ct0 + 2 * t) * 16 + lr;
#pragma unroll
      for (int v = 0; v < 8; ++v) {
        int lrow = rt * 16 + lh * 8 + v;
        int grow = rowBlk + lrow;
        float x = 0.0f;
        if (grow < M) {
          x = acc[t][v] + P[(size_t)grow * 128 + col] + b1[col];
          x = (x > 0.0f) ? x : (expf(x) - 1.0f);   // elu
        }
        H3s[lrow][col] = (_Float16)x;
      }
    }
  }
  __syncthreads();

  // ---- stage 2: 2 col-tiles per wave (ct = ct0 + 2t), K = 128 (kt = 0..3) ----
  {
    const v16h* bb = W3p + (size_t)ct0 * 32 + lane;   // ct stride 64, kt stride 4*32 = 128
    v8f acc[2] = {};
#pragma unroll
    for (int kt = 0; kt < 4; ++kt) {
      const _Float16* ap = &H3s[rt * 16 + lr][kt * 32 + lh * 8];
      v16h a = mkfrag(*(const v8h*)ap, *(const v8h*)(ap + 16));
      acc[0] = wmma16(a, bb[kt * 128],      acc[0]);
      acc[1] = wmma16(a, bb[kt * 128 + 64], acc[1]);
    }
#pragma unroll
    for (int t = 0; t < 2; ++t) {
      int col = (ct0 + 2 * t) * 16 + lr;
#pragma unroll
      for (int v = 0; v < 8; ++v) {
        int grow = rowBlk + rt * 16 + lh * 8 + v;
        if (grow < M) {
          float r = fmaxf(acc[t][v], 0.0f);         // relu
          OUT[(size_t)grow * 64 + col] = r + H4[(size_t)grow * 64 + col];
        }
      }
    }
  }
}

// ---------------------------------------------------------------------------
// Final prediction: gather rows, dot, biases, scaled sigmoid
// ---------------------------------------------------------------------------
__global__ void k_pred(const int* __restrict__ ui, const int* __restrict__ bi,
                       const float* __restrict__ U, const float* __restrict__ B,
                       const float* __restrict__ bu, const float* __restrict__ bb,
                       const float* __restrict__ bg, float* __restrict__ pred, int Q) {
  int q = blockIdx.x * blockDim.x + threadIdx.x;
  if (q >= Q) return;
  int u = ui[q], b = bi[q];
  const float* up = U + (size_t)u * 64;
  const float* bp = B + (size_t)b * 64;
  float s = 0.f;
#pragma unroll 8
  for (int k = 0; k < 64; ++k) s += up[k] * bp[k];
  float logit = s + bu[u] + bb[b] + bg[0];
  pred[q] = 4.0f / (1.0f + expf(-logit)) + 1.0f;
}

// ---------------------------------------------------------------------------
extern "C" void kernel_launch(void* const* d_in, const int* in_sizes, int n_in,
                              void* d_out, int out_size, void* d_ws, size_t ws_size,
                              hipStream_t stream) {
  (void)in_sizes; (void)n_in; (void)out_size; (void)ws_size;
  const int NU = 100000, NB = 50000, EU = 1000000, EB = 500000, Q = 100000;

  const float* S_u   = (const float*)d_in[0];
  const float* S_b   = (const float*)d_in[1];
  const int*   ei_u  = (const int*)d_in[2];
  const int*   ei_b0 = (const int*)d_in[3];
  const int*   ei_b1 = (const int*)d_in[4];
  const int*   ei_b2 = (const int*)d_in[5];
  const int*   uidx  = (const int*)d_in[6];
  const int*   bidx  = (const int*)d_in[7];
  const float* W1_u  = (const float*)d_in[8];
  const float* W1_b  = (const float*)d_in[9];
  const float* a_u   = (const float*)d_in[10];
  const float* a_b   = (const float*)d_in[11];
  const float* omega = (const float*)d_in[12];
  const float* W2_u  = (const float*)d_in[13];
  const float* W2_us = (const float*)d_in[14];
  const float* b1_u  = (const float*)d_in[15];
  const float* W2_b  = (const float*)d_in[16];
  const float* W2_bs = (const float*)d_in[17];
  const float* b1_b  = (const float*)d_in[18];
  const float* W3_u  = (const float*)d_in[19];
  const float* W3_b  = (const float*)d_in[20];
  const float* H4_u  = (const float*)d_in[21];
  const float* H4_b  = (const float*)d_in[22];
  const float* bias_u = (const float*)d_in[23];
  const float* bias_b = (const float*)d_in[24];
  const float* bias_g = (const float*)d_in[25];

  // Workspace carve-out
  char* ws = (char*)d_ws;
  size_t off = 0;
  auto carve = [&](size_t bytes) {
    char* p = ws + off;
    off = (off + bytes + 255) & ~(size_t)255;
    return p;
  };
  float* H1_u  = (float*)carve((size_t)NU * 64 * 4);
  float* H1_b  = (float*)carve((size_t)NB * 64 * 4);
  float* P_u   = (float*)carve((size_t)NU * 128 * 4);
  float* P_b   = (float*)carve((size_t)NB * 128 * 4);
  float* el_u  = (float*)carve((size_t)NU * 4);
  float* er_u  = (float*)carve((size_t)NU * 4);
  float* el_b  = (float*)carve((size_t)NB * 4);
  float* er_b  = (float*)carve((size_t)NB * 4);
  unsigned* mx_u = (unsigned*)carve((size_t)NU * 4);
  float* sm_u  = (float*)carve((size_t)NU * 4);
  float* es_u  = (float*)carve((size_t)EU * 4);
  float* H2_u  = (float*)carve((size_t)NU * 64 * 4);
  float* H2_b  = (float*)carve((size_t)NB * 64 * 4);
  unsigned* mx_b = (unsigned*)carve((size_t)NB * 4);
  float* sm_b  = (float*)carve((size_t)NB * 4);
  unsigned long long* hkeys = (unsigned long long*)carve((size_t)HSIZE * 8);
  float* hvals = (float*)carve((size_t)HSIZE * 4);
  // pre-packed WMMA B-fragment weights (tiny, L2-resident)
  const int NWC = 8 * 12 * 512;   // K=256, N=192
  const int NW2 = 2 * 8 * 512;    // K=64,  N=128
  const int NW3 = 4 * 4 * 512;    // K=128, N=64
  _Float16* WcP_u = (_Float16*)carve((size_t)NWC * 2);
  _Float16* WcP_b = (_Float16*)carve((size_t)NWC * 2);
  _Float16* W2p_u = (_Float16*)carve((size_t)NW2 * 2);
  _Float16* W3p_u = (_Float16*)carve((size_t)NW3 * 2);
  _Float16* W2p_b = (_Float16*)carve((size_t)NW2 * 2);
  _Float16* W3p_b = (_Float16*)carve((size_t)NW3 * 2);

  // Output layout: pred[Q] | U_all[NU*64] | B_all[NB*64]
  float* pred  = (float*)d_out;
  float* U_all = pred + Q;
  float* B_all = U_all + (size_t)NU * 64;

  const unsigned NEGINF_ORD = 0x007FFFFFu;  // ford(-inf)
  auto blk = [](long long n) { return (unsigned)((n + 255) / 256); };

  // --- init accumulators ---
  k_fill_u32<<<blk(NU), 256, 0, stream>>>(mx_u, NEGINF_ORD, NU);
  k_fill_f32<<<blk(NU), 256, 0, stream>>>(sm_u, 0.f, NU);
  k_fill_f32<<<blk((long long)NU * 64), 256, 0, stream>>>(H2_u, 0.f, (long long)NU * 64);
  k_fill_u32<<<blk(NB), 256, 0, stream>>>(mx_b, NEGINF_ORD, NB);
  k_fill_f32<<<blk(NB), 256, 0, stream>>>(sm_b, 0.f, NB);
  k_fill_f32<<<blk((long long)NB * 64), 256, 0, stream>>>(H2_b, 0.f, (long long)NB * 64);
  k_fill_u64<<<blk(HSIZE), 256, 0, stream>>>(hkeys, EMPTYKEY, (long long)HSIZE);
  k_fill_f32<<<blk(HSIZE), 256, 0, stream>>>(hvals, 0.f, (long long)HSIZE);

  // --- pack weights into WMMA B-fragment layout ---
  k_pack_b<<<blk(NWC), 256, 0, stream>>>(W1_u, 64, W2_us, 128, WcP_u, 256);
  k_pack_b<<<blk(NWC), 256, 0, stream>>>(W1_b, 64, W2_bs, 128, WcP_b, 256);
  k_pack_b<<<blk(NW2), 256, 0, stream>>>(W2_u, 128, W2_u, 0, W2p_u, 64);
  k_pack_b<<<blk(NW3), 256, 0, stream>>>(W3_u, 64, W3_u, 0, W3p_u, 128);
  k_pack_b<<<blk(NW2), 256, 0, stream>>>(W2_b, 128, W2_b, 0, W2p_b, 64);
  k_pack_b<<<blk(NW3), 256, 0, stream>>>(W3_b, 64, W3_b, 0, W3p_b, 128);

  // --- layer 1: fused S@[W1 | W2s] (single pass over S), WMMA f16 ---
  k_gemm1<<<(NU + 63) / 64, 256, 0, stream>>>(S_u, WcP_u, H1_u, P_u, NU);
  k_gemm1<<<(NB + 63) / 64, 256, 0, stream>>>(S_b, WcP_b, H1_b, P_b, NB);
  k_elr<<<blk(NU), 256, 0, stream>>>(H1_u, a_u, el_u, er_u, NU);
  k_elr<<<blk(NB), 256, 0, stream>>>(H1_b, a_b, el_b, er_b, NB);

  // --- user graph attention ---
  k_edge_pass1<<<blk(EU), 256, 0, stream>>>(ei_u, EU, el_u, er_u, es_u, mx_u);
  k_edge_pass2<<<blk(EU), 256, 0, stream>>>(ei_u, EU, es_u, mx_u, sm_u);
  k_edge_scatter<<<blk((long long)EU * 64), 256, 0, stream>>>(ei_u, EU, es_u, sm_u, H1_u, H2_u);

  // --- biz multi-graph attention with duplicate-edge merge ---
  k_hash_insert<<<blk((long long)3 * EB), 256, 0, stream>>>(
      ei_b0, ei_b1, ei_b2, EB, el_b, er_b, omega, hkeys, hvals);
  k_hash_max<<<blk(HSIZE), 256, 0, stream>>>(hkeys, hvals, mx_b);
  k_hash_exp<<<blk(HSIZE), 256, 0, stream>>>(hkeys, hvals, mx_b, sm_b);
  k_hash_scatter<<<blk(HSIZE), 256, 0, stream>>>(hkeys, hvals, sm_b, H1_b, H2_b);

  // --- layer 2 + latent factors (fused double-WMMA kernel) ---
  k_layer2<<<(NU + 63) / 64, 256, 0, stream>>>(H2_u, P_u, b1_u, W2p_u, W3p_u, H4_u, U_all, NU);
  k_layer2<<<(NB + 63) / 64, 256, 0, stream>>>(H2_b, P_b, b1_b, W2p_b, W3p_b, H4_b, B_all, NB);

  // --- prediction ---
  k_pred<<<blk(Q), 256, 0, stream>>>(uidx, bidx, U_all, B_all, bias_u, bias_b,
                                     bias_g, pred, Q);
}
